// WaveNet_39118562132426
// MI455X (gfx1250) — matrix operand order repaired and checked by reading, not compiled, not run on hardware
//
#include <hip/hip_runtime.h>
#include <hip/hip_bf16.h>

// ---------------------------------------------------------------------------
// WaveNet on MI455X (gfx1250): all 64x64 GEMMs via v_wmma_f32_16x16x32_bf16
// (wave32). Weights pre-packed ONCE into bf16 LDS-fragment images in d_ws;
// each block fetches its 48KB weight image with the Tensor Data Mover
// (tensor_load_to_lds, TENSORcnt) overlapped with activation fragment build.
// ---------------------------------------------------------------------------

typedef __bf16 bf16;
typedef bf16     v16bf __attribute__((ext_vector_type(16)));
typedef float    v8f   __attribute__((ext_vector_type(8)));
typedef unsigned v4u   __attribute__((ext_vector_type(4)));
typedef int      v4i   __attribute__((ext_vector_type(4)));
typedef int      v8i   __attribute__((ext_vector_type(8)));

#define T_LEN   65536
#define W_OUT   63489          // 65536 - 2047
#define TILE_T  128

// ---- LDS layout: layer kernel -------------------------------------------
#define L_LW1   0              // stage1 weights [8mt][4ks] frags  : 32768 B
#define L_LW2   32768          // stage2 weights [8mt][2ks] frags  : 16384 B
#define L_WBYTES 49152         // contiguous packed weight image
#define L_LX    49152          // X frags        [4ks][8nt]        : 32768 B
#define L_LZ    81920          // z frags        [2ks][8nt]        : 16384 B
#define L_BIAS  98304          // bf,bg,br,bs (4*64 f32)           :  1024 B
#define L_SIZE  99328

// ---- LDS layout: head kernel --------------------------------------------
#define O_W0    0              // [4mt][2ks]   :  8192 B
#define O_W1    8192           // [16mt][2ks]  : 32768 B
#define O_WBYTES 40960         // contiguous packed weight image
#define O_SX    40960          // [2ks][8nt]   : 16384 B
#define O_MID   57344          // [2ks][8nt]   : 16384 B
#define O_CST   73728          // a0,a1,b0 (64 each) + b1 (256)  : 1792 B
#define O_SIZE  75520

// ---- d_ws layout ----------------------------------------------------------
#define WS_HA    0
#define WS_HB    ((size_t)64 * T_LEN * 4)
#define WS_SKIP  ((size_t)128 * T_LEN * 4)
#define WS_WPACK (WS_SKIP + (size_t)64 * W_OUT * 4)
#define LAYER_PACK_ELTS 24576   // 49152 B / 2 per layer

#define HAVE_TDM __has_builtin(__builtin_amdgcn_tensor_load_to_lds)

// --- compile-time diagnostics (visible in hipcc stderr) --------------------
#if HAVE_TDM
#pragma message("CDNA5-DIAG: tensor_load_to_lds builtin AVAILABLE in this pass")
#else
#pragma message("CDNA5-DIAG: tensor_load_to_lds builtin NOT available in this pass (b128 fallback)")
#endif
#if defined(__HIP_DEVICE_COMPILE__)
#pragma message("CDNA5-DIAG: this is the DEVICE pass")
#endif
#if __clang_major__ >= 23
#pragma message("CDNA5-DIAG: clang >= 23 (6-arg TDM builtin)")
#else
#pragma message("CDNA5-DIAG: clang < 23 (5-arg TDM builtin)")
#endif

__device__ __forceinline__ v8f wmma_bf16(v16bf a, v16bf b, v8f c) {
  return __builtin_amdgcn_wmma_f32_16x16x32_bf16(false, a, false, b,
                                                 (short)0, c, false, false);
}

__device__ __forceinline__ const v16bf* frag_ptr(const char* sm, int base,
                                                 int frag, int lane) {
  return reinterpret_cast<const v16bf*>(sm + base + frag * 1024 + lane * 32);
}

// Issue one linear TDM copy (nbytes, 8-byte elements) global -> LDS.
// D# group0: count=1, lds_addr, global_addr, type=2. group1: data_size=8B,
// tensor_dim0 = tile_dim0 = n8, tensor_dim1 = tile_dim1 = 1, stride0 = n8.
__device__ __forceinline__ void tdm_load_linear(void* lds_dst, const void* gsrc,
                                                unsigned nbytes) {
#if HAVE_TDM
  unsigned lds = (unsigned)(uintptr_t)lds_dst;   // flat addr low bits == LDS addr
  unsigned long long ga = (unsigned long long)(uintptr_t)gsrc;
  unsigned n8 = nbytes >> 3;                     // < 65536 (fits tile_dim0)
  v4u g0;
  g0.x = 1u;                                           // count = 1
  g0.y = lds;                                          // lds_addr
  g0.z = (unsigned)(ga & 0xffffffffu);                 // global_addr[31:0]
  g0.w = (unsigned)((ga >> 32) & 0x01ffffffu) | (2u << 30);  // [56:32] | type=2
  v8i g1;
  g1[0] = (int)(3u << 16);                             // data_size = 8B
  g1[1] = (int)((n8 & 0xffffu) << 16);                 // tensor_dim0[15:0]
  g1[2] = (int)((n8 >> 16) | (1u << 16));              // td0[31:16] | tensor_dim1=1
  g1[3] = (int)((n8 & 0xffffu) << 16);                 // tile_dim0 = n8
  g1[4] = 1;                                           // tile_dim1 = 1
  g1[5] = (int)n8;                                     // tensor_dim0_stride lo
  g1[6] = 0;
  g1[7] = 0;
  v4i gz = {0, 0, 0, 0};
#if __clang_major__ >= 23
  v8i gz8 = {0, 0, 0, 0, 0, 0, 0, 0};
  __builtin_amdgcn_tensor_load_to_lds(g0, g1, gz, gz, gz8, 0);
#else
  __builtin_amdgcn_tensor_load_to_lds(g0, g1, gz, gz, 0);
#endif
#else
  (void)lds_dst; (void)gsrc; (void)nbytes;
#endif
}

__device__ __forceinline__ void lds_copy_fallback(char* lds_dst,
                                                  const char* gsrc,
                                                  unsigned nbytes, int tid) {
  const uint4* s = (const uint4*)gsrc;
  uint4* d = (uint4*)lds_dst;
  for (unsigned i = tid; i < (nbytes >> 4); i += 256) d[i] = s[i];
}

// ---------------------------------------------------------------------------
__global__ void __launch_bounds__(256) init_h(float* __restrict__ h,
                                              const float* __restrict__ input,
                                              const float* __restrict__ w0,
                                              const float* __restrict__ b0) {
  int idx = blockIdx.x * 256 + threadIdx.x;   // idx == c*T + t, T = 65536
  int c = idx >> 16, t = idx & 65535;
  h[idx] = w0[c] * input[t] + b0[c];
}

__global__ void __launch_bounds__(256) zero_skip(float* __restrict__ s, int n) {
  int i = blockIdx.x * 256 + threadIdx.x;
  if (i < n) s[i] = 0.0f;
}

// ---------------------------------------------------------------------------
// Pre-pack all weights into bf16 fragment images (exact LDS layout).
// Blocks 0..19: layers. Block 20: output head.
// ---------------------------------------------------------------------------
__global__ void __launch_bounds__(256) pack_weights(
    const float* __restrict__ wf, const float* __restrict__ wg,
    const float* __restrict__ wr, const float* __restrict__ wsm,
    const float* __restrict__ wo0, const float* __restrict__ wo1,
    bf16* __restrict__ wpack) {
  const int b = blockIdx.x, tid = threadIdx.x;
  if (b < 20) {
    bf16* dst = wpack + (size_t)b * LAYER_PACK_ELTS;
    const float* wfl = wf + (size_t)b * 8192;
    const float* wgl = wg + (size_t)b * 8192;
    const float* wrl = wr + (size_t)b * 4096;
    const float* wsl = wsm + (size_t)b * 4096;
    // stage-1: Wcat(128x128) = [[Wf0 Wf1],[Wg0 Wg1]]  (A-operand layout)
    for (int idx = tid; idx < 16384; idx += 256) {
      int lf = idx & 31, e = (idx >> 5) & 15, frag = idx >> 9;  // mt*4+ks
      int ks = frag & 3, mt = frag >> 2;
      int m = 16 * mt + (lf & 15);
      int K = 32 * ks + 16 * (lf >> 4) + e;
      int half = K >> 6, c = K & 63;
      float v = (m < 64) ? wfl[(m * 64 + c) * 2 + half]
                         : wgl[((m - 64) * 64 + c) * 2 + half];
      dst[frag * 512 + lf * 16 + e] = (bf16)v;
    }
    // stage-2: Wrs(128x64) = [Wr; Ws]
    for (int idx = tid; idx < 8192; idx += 256) {
      int lf = idx & 31, e = (idx >> 5) & 15, frag = idx >> 9;  // mt*2+ks
      int ks = frag & 1, mt = frag >> 1;
      int m = 16 * mt + (lf & 15);
      int K = 32 * ks + 16 * (lf >> 4) + e;
      float v = (m < 64) ? wrl[m * 64 + K] : wsl[(m - 64) * 64 + K];
      dst[16384 + frag * 512 + lf * 16 + e] = (bf16)v;
    }
  } else {
    bf16* dst = wpack + (size_t)20 * LAYER_PACK_ELTS;
    for (int idx = tid; idx < 4096; idx += 256) {        // W_out0 (64x64)
      int lf = idx & 31, e = (idx >> 5) & 15, frag = idx >> 9;
      int ks = frag & 1, mt = frag >> 1;
      int m = 16 * mt + (lf & 15);
      int K = 32 * ks + 16 * (lf >> 4) + e;
      dst[frag * 512 + lf * 16 + e] = (bf16)wo0[m * 64 + K];
    }
    for (int idx = tid; idx < 16384; idx += 256) {       // W_out1 (256x64)
      int lf = idx & 31, e = (idx >> 5) & 15, frag = idx >> 9;
      int ks = frag & 1, mt = frag >> 1;
      int m = 16 * mt + (lf & 15);
      int K = 32 * ks + 16 * (lf >> 4) + e;
      dst[4096 + frag * 512 + lf * 16 + e] = (bf16)wo1[m * 64 + K];
    }
  }
}

// ---------------------------------------------------------------------------
// One dilated-gated-residual layer (global time coords, tile of 128).
// ---------------------------------------------------------------------------
__global__ void __launch_bounds__(256) wavenet_layer(
    const float* __restrict__ h_cur, float* __restrict__ h_next,
    const bf16* __restrict__ wl,     // packed fragment image, 49152 B
    const float* __restrict__ bfv, const float* __restrict__ bgv,
    const float* __restrict__ brv, const float* __restrict__ bsv,
    float* __restrict__ skip, int d, int gstart) {
  extern __shared__ char sm[];
  const int tid  = threadIdx.x;
  const int lane = tid & 31;
  const int wv   = tid >> 5;
  const int g0   = gstart + blockIdx.x * TILE_T;

#if HAVE_TDM
  if (wv == 0) tdm_load_linear(sm + L_LW1, wl, L_WBYTES);  // async weight DMA
#else
  lds_copy_fallback(sm + L_LW1, (const char*)wl, L_WBYTES, tid);
#endif

  // --- X fragments (B op): rows<64 -> left = h[g-d], rows>=64 -> right = h[g]
  for (int idx = tid; idx < 4 * 8 * 512; idx += 256) {
    int lf = idx & 31, e = (idx >> 5) & 15, frag = idx >> 9;  // frag = ks*8+nt
    int ks = frag >> 3, nt = frag & 7;
    int K = 32 * ks + 16 * (lf >> 4) + e;                     // B layout
    int g = g0 + 16 * nt + (lf & 15);
    if (g > T_LEN - 1) g = T_LEN - 1;                         // clamp tail tile
    float v = (K < 64) ? h_cur[K * T_LEN + (g - d)]
                       : h_cur[(K - 64) * T_LEN + g];
    ((bf16*)(sm + L_LX))[frag * 512 + lf * 16 + e] = (bf16)v;
  }
  if (tid < 64) {
    float* bb = (float*)(sm + L_BIAS);
    bb[tid] = bfv[tid]; bb[64 + tid] = bgv[tid];
    bb[128 + tid] = brv[tid]; bb[192 + tid] = bsv[tid];
  }
#if HAVE_TDM
  if (wv == 0) __builtin_amdgcn_s_wait_tensorcnt(0);  // DMA done (wave 0 only)
#endif
  __syncthreads();

  // --- stage 1: [xf;xg](128 x 128) = Wcat @ X; wave owns matching xf/xg rows
  const v8f vzero = {0.f, 0.f, 0.f, 0.f, 0.f, 0.f, 0.f, 0.f};
  const int mta = wv & 3;             // xf M-tile; xg M-tile = mta+4 (same rows)
  const int ntb = (wv >> 2) * 4;      // 4 N-tiles per wave
  v8f accF[4] = {vzero, vzero, vzero, vzero};
  v8f accG[4] = {vzero, vzero, vzero, vzero};
#pragma unroll
  for (int ks = 0; ks < 4; ++ks) {
    v16bf a0 = *frag_ptr(sm, L_LW1, mta * 4 + ks, lane);
    v16bf a1 = *frag_ptr(sm, L_LW1, (mta + 4) * 4 + ks, lane);
#pragma unroll
    for (int nt = 0; nt < 4; ++nt) {
      v16bf b = *frag_ptr(sm, L_LX, ks * 8 + ntb + nt, lane);
      accF[nt] = wmma_bf16(a0, b, accF[nt]);
      accG[nt] = wmma_bf16(a1, b, accG[nt]);
    }
  }
  // gated activation in registers -> z fragments (B layout for stage 2)
  {
    const float* bfp = (const float*)(sm + L_BIAS);
    const float* bgp = bfp + 64;
#pragma unroll
    for (int nt = 0; nt < 4; ++nt) {
#pragma unroll
      for (int j = 0; j < 8; ++j) {
        int rz = 16 * mta + j + 8 * (lane >> 4);  // z row 0..63
        float xf = accF[nt][j] + bfp[rz];
        float xg = accG[nt][j] + bgp[rz];
        float z  = tanhf(xf) * (1.0f / (1.0f + __expf(-xg)));
        int laneP = (lane & 15) + 16 * ((rz >> 4) & 1);
        int frag  = (rz >> 5) * 8 + (ntb + nt);
        ((bf16*)(sm + L_LZ))[frag * 512 + laneP * 16 + (rz & 15)] = (bf16)z;
      }
    }
  }
  __syncthreads();

  // --- stage 2: [res;skip](128 x 128) = [Wr;Ws] @ z
  v8f acc2[8];
#pragma unroll
  for (int nt = 0; nt < 8; ++nt) acc2[nt] = vzero;
#pragma unroll
  for (int ks = 0; ks < 2; ++ks) {
    v16bf a = *frag_ptr(sm, L_LW2, wv * 2 + ks, lane);
#pragma unroll
    for (int nt = 0; nt < 8; ++nt) {
      v16bf b = *frag_ptr(sm, L_LZ, ks * 8 + nt, lane);
      acc2[nt] = wmma_bf16(a, b, acc2[nt]);
    }
  }
  // epilogue: residual update (waves 0-3) / skip accumulation (waves 4-7)
  {
    const float* brp = (const float*)(sm + L_BIAS) + 128;
    const float* bsp = brp + 64;
#pragma unroll
    for (int nt = 0; nt < 8; ++nt) {
#pragma unroll
      for (int j = 0; j < 8; ++j) {
        int rit = j + 8 * (lane >> 4);
        int g = g0 + 16 * nt + (lane & 15);
        if (wv < 4) {
          int o = 16 * wv + rit;
          if (g < T_LEN)
            h_next[o * T_LEN + g] = acc2[nt][j] + brp[o] + h_cur[o * T_LEN + g];
        } else {
          int srow = 16 * (wv - 4) + rit;
          if (g >= 2047 && g < T_LEN) {
            size_t off = (size_t)srow * W_OUT + (g - 2047);
            skip[off] += acc2[nt][j] + bsp[srow];
          }
        }
      }
    }
  }
}

// ---------------------------------------------------------------------------
// Output head: leaky(a0) -> W0(64x64)+b0 -> leaky(a1) -> W1(256x64)+b1
// ---------------------------------------------------------------------------
__global__ void __launch_bounds__(256) wavenet_head(
    const float* __restrict__ skip, const float* __restrict__ alpha,
    const bf16* __restrict__ wh,     // packed fragment image, 40960 B
    const float* __restrict__ b0, const float* __restrict__ b1,
    float* __restrict__ out) {
  extern __shared__ char sm[];
  const int tid = threadIdx.x, lane = tid & 31, wv = tid >> 5;
  const int t0 = blockIdx.x * TILE_T;

#if HAVE_TDM
  if (wv == 0) tdm_load_linear(sm + O_W0, wh, O_WBYTES);
#else
  lds_copy_fallback(sm + O_W0, (const char*)wh, O_WBYTES, tid);
#endif

  float* cst = (float*)(sm + O_CST);
  if (tid < 64) {
    cst[64 + tid] = alpha[64 + tid];  // alpha1
    cst[128 + tid] = b0[tid];
  }
  cst[192 + tid] = b1[tid];
  // skip fragments (B) with first leaky (alpha0 read from global: pre-barrier)
  for (int idx = tid; idx < 2 * 8 * 512; idx += 256) {
    int lf = idx & 31, e = (idx >> 5) & 15, frag = idx >> 9;
    int ks = frag >> 3, nt = frag & 7;
    int s = 32 * ks + 16 * (lf >> 4) + e;
    int t = t0 + 16 * nt + (lf & 15);
    float v = 0.0f;
    if (t < W_OUT) {
      v = skip[(size_t)s * W_OUT + t];
      float a = alpha[s];
      v = (v > 0.0f) ? v : a * v;
    }
    ((bf16*)(sm + O_SX))[frag * 512 + lf * 16 + e] = (bf16)v;
  }
#if HAVE_TDM
  if (wv == 0) __builtin_amdgcn_s_wait_tensorcnt(0);
#endif
  __syncthreads();

  // GEMM1: mid(64 x 128) = W0 @ sx
  const v8f vzero = {0.f, 0.f, 0.f, 0.f, 0.f, 0.f, 0.f, 0.f};
  const int mta = wv & 3, ntb = (wv >> 2) * 4;
  v8f acc1[4] = {vzero, vzero, vzero, vzero};
#pragma unroll
  for (int ks = 0; ks < 2; ++ks) {
    v16bf a = *frag_ptr(sm, O_W0, mta * 2 + ks, lane);
#pragma unroll
    for (int nt = 0; nt < 4; ++nt) {
      v16bf b = *frag_ptr(sm, O_SX, ks * 8 + ntb + nt, lane);
      acc1[nt] = wmma_bf16(a, b, acc1[nt]);
    }
  }
#pragma unroll
  for (int nt = 0; nt < 4; ++nt) {
#pragma unroll
    for (int j = 0; j < 8; ++j) {
      int rz = 16 * mta + j + 8 * (lane >> 4);
      float v = acc1[nt][j] + cst[128 + rz];
      v = (v > 0.0f) ? v : cst[64 + rz] * v;
      int laneP = (lane & 15) + 16 * ((rz >> 4) & 1);
      int frag  = (rz >> 5) * 8 + (ntb + nt);
      ((bf16*)(sm + O_MID))[frag * 512 + laneP * 16 + (rz & 15)] = (bf16)v;
    }
  }
  __syncthreads();

  // GEMM2: out(256 x 128) = W1 @ mid; wave owns M-tiles {wv, wv+8}
  v8f acc2a[8], acc2b[8];
#pragma unroll
  for (int nt = 0; nt < 8; ++nt) { acc2a[nt] = vzero; acc2b[nt] = vzero; }
#pragma unroll
  for (int ks = 0; ks < 2; ++ks) {
    v16bf a0 = *frag_ptr(sm, O_W1, wv * 2 + ks, lane);
    v16bf a1 = *frag_ptr(sm, O_W1, (wv + 8) * 2 + ks, lane);
#pragma unroll
    for (int nt = 0; nt < 8; ++nt) {
      v16bf b = *frag_ptr(sm, O_MID, ks * 8 + nt, lane);
      acc2a[nt] = wmma_bf16(a0, b, acc2a[nt]);
      acc2b[nt] = wmma_bf16(a1, b, acc2b[nt]);
    }
  }
#pragma unroll
  for (int nt = 0; nt < 8; ++nt) {
    int t = t0 + 16 * nt + (lane & 15);
    if (t >= W_OUT) continue;
#pragma unroll
    for (int j = 0; j < 8; ++j) {
      int rit = j + 8 * (lane >> 4);
      int oa = 16 * wv + rit;
      int ob = 16 * (wv + 8) + rit;
      out[(size_t)oa * W_OUT + t] = acc2a[nt][j] + cst[192 + oa];
      out[(size_t)ob * W_OUT + t] = acc2b[nt][j] + cst[192 + ob];
    }
  }
}

// ---------------------------------------------------------------------------
extern "C" void kernel_launch(void* const* d_in, const int* in_sizes, int n_in,
                              void* d_out, int out_size, void* d_ws,
                              size_t ws_size, hipStream_t stream) {
  const float* input = (const float*)d_in[0];
  const float* w0v   = (const float*)d_in[1];
  const float* b0v   = (const float*)d_in[2];
  const float* wf    = (const float*)d_in[3];
  const float* bfv   = (const float*)d_in[4];
  const float* wg    = (const float*)d_in[5];
  const float* bgv   = (const float*)d_in[6];
  const float* wr    = (const float*)d_in[7];
  const float* brv   = (const float*)d_in[8];
  const float* wsv   = (const float*)d_in[9];
  const float* bsv   = (const float*)d_in[10];
  const float* alpha = (const float*)d_in[11];
  const float* wo0   = (const float*)d_in[12];
  const float* bo0   = (const float*)d_in[13];
  const float* wo1   = (const float*)d_in[14];
  const float* bo1   = (const float*)d_in[15];

  char* wsb    = (char*)d_ws;
  float* hA    = (float*)(wsb + WS_HA);      // 16.78 MB
  float* hB    = (float*)(wsb + WS_HB);      // 16.78 MB
  float* skip  = (float*)(wsb + WS_SKIP);    // 16.25 MB
  bf16*  wpack = (bf16*)(wsb + WS_WPACK);    // 1.00 MB packed weights

  pack_weights<<<21, 256, 0, stream>>>(wf, wg, wr, wsv, wo0, wo1, wpack);
  init_h<<<(64 * T_LEN) / 256, 256, 0, stream>>>(hA, input, w0v, b0v);
  zero_skip<<<(64 * W_OUT + 255) / 256, 256, 0, stream>>>(skip, 64 * W_OUT);

  float* hc = hA;
  float* hn = hB;
  int shift = 0;
  for (int i = 0; i < 20; ++i) {
    int d = 1 << (i % 10);
    int gstart = shift + d;
    int nb = (T_LEN - gstart + TILE_T - 1) / TILE_T;
    wavenet_layer<<<nb, 256, L_SIZE, stream>>>(
        hc, hn, wpack + (size_t)i * LAYER_PACK_ELTS,
        bfv + i * 64, bgv + i * 64, brv + i * 64, bsv + i * 64,
        skip, d, gstart);
    float* tmp = hc; hc = hn; hn = tmp;
    shift = gstart;
  }

  wavenet_head<<<(W_OUT + TILE_T - 1) / TILE_T, 256, O_SIZE, stream>>>(
      skip, alpha, wpack + (size_t)20 * LAYER_PACK_ELTS, bo0, bo1,
      (float*)d_out);
}
